// Mix_Attn_Block_73254962200821
// MI455X (gfx1250) — matrix operand order repaired and checked by reading, not compiled, not run on hardware
//
#include <hip/hip_runtime.h>
#include <stdint.h>
#include <stddef.h>

typedef __bf16 bf16;
typedef __attribute__((ext_vector_type(16))) __bf16 v16bf;
typedef __attribute__((ext_vector_type(8)))  __bf16 v8bf;
typedef __attribute__((ext_vector_type(8)))  float  v8f;
typedef int v4i __attribute__((vector_size(16)));   // matches builtin param type

#define BB      8
#define LSEQ    1024
#define DMODEL  1024
#define NHEAD   16
#define DH      64
#define MROWS   (BB * LSEQ)   // 8192

#if defined(__has_builtin)
#if __has_builtin(__builtin_amdgcn_global_load_async_to_lds_b128) && \
    __has_builtin(__builtin_amdgcn_s_wait_asynccnt)
#define ASYNC_OK 1
#endif
#endif
#ifndef ASYNC_OK
#define ASYNC_OK 0
#endif

__device__ __forceinline__ bf16 f2bf(float f) {
  union { float f; uint32_t u; } c; c.f = f;
  uint32_t r = c.u + 0x7FFFu + ((c.u >> 16) & 1u);   // round-to-nearest-even
  union { uint16_t s; bf16 b; } o; o.s = (uint16_t)(r >> 16);
  return o.b;
}

// 16-byte global -> LDS copy; async (ASYNCcnt-tracked) when available.
__device__ __forceinline__ void cp16(void* lds, const void* g) {
#if ASYNC_OK
  __builtin_amdgcn_global_load_async_to_lds_b128(
      (__attribute__((address_space(1))) v4i*)g,
      (__attribute__((address_space(3))) v4i*)lds, 0, 0);
#else
  *(v8bf*)lds = *(const v8bf*)g;
#endif
}

// ---------------------------------------------------------------------------
// Kernel 1: fp32 weight [K][N] -> bf16 transposed [N][K]
// grid: (1024, 6), block: 256
// ---------------------------------------------------------------------------
__global__ __launch_bounds__(256) void wcvt_kernel(
    const float* __restrict__ W0, const float* __restrict__ W1,
    const float* __restrict__ W2, const float* __restrict__ W3,
    const float* __restrict__ W4, const float* __restrict__ W5,
    bf16* __restrict__ T0, bf16* __restrict__ T1, bf16* __restrict__ T2,
    bf16* __restrict__ T3, bf16* __restrict__ T4, bf16* __restrict__ T5) {
  const float* W; bf16* T;
  switch (blockIdx.y) {
    case 0: W = W0; T = T0; break;
    case 1: W = W1; T = T1; break;
    case 2: W = W2; T = T2; break;
    case 3: W = W3; T = T3; break;
    case 4: W = W4; T = T4; break;
    default: W = W5; T = T5; break;
  }
  int e = (blockIdx.x * 256 + threadIdx.x) * 4;
#pragma unroll
  for (int i = 0; i < 4; ++i) {
    int idx = e + i;
    int n = idx >> 10, k = idx & 1023;
    T[idx] = f2bf(W[k * DMODEL + n]);          // T[n*1024 + k] = W[k][n]
  }
}

// ---------------------------------------------------------------------------
// Kernel 2: LayerNorm fp32 -> bf16.  grid: (8192, 2), block: 256
// ---------------------------------------------------------------------------
__global__ __launch_bounds__(256) void ln_kernel(
    const float* __restrict__ a, const float* __restrict__ b,
    const float* __restrict__ ga, const float* __restrict__ ba,
    const float* __restrict__ gb, const float* __restrict__ bb,
    bf16* __restrict__ oa, bf16* __restrict__ ob) {
  const int row = blockIdx.x, which = blockIdx.y, tid = threadIdx.x;
  const float* x  = (which ? b : a) + (long)row * DMODEL;
  const float* g  = which ? gb : ga;
  const float* be = which ? bb : ba;
  bf16* o = (which ? ob : oa) + (long)row * DMODEL;

  __shared__ float red[256];
  float v[4], s = 0.f;
#pragma unroll
  for (int i = 0; i < 4; ++i) { v[i] = x[tid + i * 256]; s += v[i]; }
  red[tid] = s; __syncthreads();
  for (int off = 128; off; off >>= 1) {
    if (tid < off) red[tid] += red[tid + off];
    __syncthreads();
  }
  float mu = red[0] * (1.0f / DMODEL);
  float s2 = 0.f;
#pragma unroll
  for (int i = 0; i < 4; ++i) { float d = v[i] - mu; s2 += d * d; }
  __syncthreads();
  red[tid] = s2; __syncthreads();
  for (int off = 128; off; off >>= 1) {
    if (tid < off) red[tid] += red[tid + off];
    __syncthreads();
  }
  float r = rsqrtf(red[0] * (1.0f / DMODEL) + 1e-5f);
#pragma unroll
  for (int i = 0; i < 4; ++i) {
    int c = tid + i * 256;
    o[c] = f2bf((v[i] - mu) * r * g[c] + be[c]);
  }
}

// ---------------------------------------------------------------------------
// Kernel 3: GEMM  Y = X(bf16, MxK) @ W (via Wt bf16 [N][K]) + bias
//   F32OUT=0: Y -> bf16.   F32OUT=1: Y+res -> fp32 out.
// block tile 128x128, BK=32, 8 waves (4x2), wave tile 32x64.
// Double-buffered LDS, filled with GLOBAL_LOAD_ASYNC_TO_LDS_B128.
// grid: (M/128, N/128), block: 256
// ---------------------------------------------------------------------------
template <bool F32OUT>
__global__ __launch_bounds__(256) void gemm_kernel(
    const bf16* __restrict__ X, const bf16* __restrict__ Wt,
    const float* __restrict__ bias, const float* __restrict__ res,
    bf16* __restrict__ outh, float* __restrict__ outf, int M, int N, int Kd) {
  __shared__ __align__(16) bf16 Xs[2][128 * 32];
  __shared__ __align__(16) bf16 Ws[2][128 * 32];

  const int tid  = threadIdx.x;
  const int lane = tid & 31;
  const int wave = tid >> 5;
  const int wm   = wave & 3;          // 0..3 along M
  const int wn   = wave >> 2;         // 0..1 along N
  const int m0   = blockIdx.x * 128;
  const int n0   = blockIdx.y * 128;

  v8f acc[2][4] = {};

  const int lr = tid >> 1;            // 0..127
  const int lg = (tid & 1) * 16;      // 0 or 16
  const bf16* xsrc = X  + (long)(m0 + lr) * Kd + lg;
  const bf16* wsrc = Wt + (long)(n0 + lr) * Kd + lg;

  // stage one 128x32 X tile + 128x32 Wt tile into buffer `buf` (4 async ops/thread)
  auto stage = [&](int buf, int k0) {
    cp16(&Xs[buf][lr * 32 + lg],     xsrc + k0);
    cp16(&Xs[buf][lr * 32 + lg + 8], xsrc + k0 + 8);
    cp16(&Ws[buf][lr * 32 + lg],     wsrc + k0);
    cp16(&Ws[buf][lr * 32 + lg + 8], wsrc + k0 + 8);
  };

  const int NT = Kd / 32;
  stage(0, 0);

  for (int kt = 0; kt < NT; ++kt) {
    const int cur = kt & 1;
    if (kt + 1 < NT) {
      stage(cur ^ 1, (kt + 1) * 32);        // prefetch next tile (4 newest ops)
#if ASYNC_OK
      __builtin_amdgcn_s_wait_asynccnt(4);  // drain current tile's older ops
#endif
    } else {
#if ASYNC_OK
      __builtin_amdgcn_s_wait_asynccnt(0);
#endif
    }
    __syncthreads();                        // publish current tile to all waves

    v16bf af[2], bfv[4];
    const int kb  = (lane >> 4) * 8;    // A-frag K group
    const int kbb = (lane >> 4) * 8;    // B-frag K group
#pragma unroll
    for (int mi = 0; mi < 2; ++mi) {
      const bf16* xr = &Xs[cur][(wm * 32 + mi * 16 + (lane & 15)) * 32 + kb];
      ((v8bf*)&af[mi])[0] = *(const v8bf*)(xr);
      ((v8bf*)&af[mi])[1] = *(const v8bf*)(xr + 16);
    }
#pragma unroll
    for (int ni = 0; ni < 4; ++ni) {
      const bf16* wr = &Ws[cur][(wn * 64 + ni * 16 + (lane & 15)) * 32 + kbb];
      ((v8bf*)&bfv[ni])[0] = *(const v8bf*)(wr);
      ((v8bf*)&bfv[ni])[1] = *(const v8bf*)(wr + 16);
    }
#pragma unroll
    for (int mi = 0; mi < 2; ++mi)
#pragma unroll
      for (int ni = 0; ni < 4; ++ni)
        acc[mi][ni] = __builtin_amdgcn_wmma_f32_16x16x32_bf16(
            false, af[mi], false, bfv[ni], (short)0, acc[mi][ni], false, false);
    __syncthreads();                        // all reads done before buffer reuse
  }

  // epilogue
  int   col[4];
  float bia[4];
#pragma unroll
  for (int ni = 0; ni < 4; ++ni) {
    col[ni] = n0 + wn * 64 + ni * 16 + (lane & 15);
    bia[ni] = bias[col[ni]];
  }
#pragma unroll
  for (int mi = 0; mi < 2; ++mi) {
    const int rb = m0 + wm * 32 + mi * 16 + (lane >> 4) * 8;
#pragma unroll
    for (int ni = 0; ni < 4; ++ni) {
#pragma unroll
      for (int i = 0; i < 8; ++i) {
        float v = acc[mi][ni][i] + bia[ni];
        long idx = (long)(rb + i) * N + col[ni];
        if (F32OUT) outf[idx] = v + res[idx];
        else        outh[idx] = f2bf(v);
      }
    }
  }
}

// ---------------------------------------------------------------------------
// Kernel 4: flash attention (one direction).
//   O[b, q, h*64+dh] = softmax_k( Q_h(q)·K_h(k)^T / 8 ) @ V_h
// grid: (L/64, H, B), block: 128 (4 waves; each wave owns 16 q-rows)
// ---------------------------------------------------------------------------
__global__ __launch_bounds__(128) void attn_kernel(
    const bf16* __restrict__ Q, const bf16* __restrict__ K,
    const bf16* __restrict__ V, bf16* __restrict__ O) {
  __shared__ __align__(16) bf16  Vt[DH * 64];      // transposed V tile [dh][key]
  __shared__ __align__(16) float Sbuf[64 * 64];
  __shared__ __align__(16) bf16  Pbuf[64 * 64];
  __shared__ float rowmax[64], rowsum[64], rowscale[64];

  const int tid  = threadIdx.x;
  const int lane = tid & 31;
  const int wave = tid >> 5;                       // 0..3
  const int q0      = blockIdx.x * 64;
  const int colbase = blockIdx.y * DH;
  const long rowbase = (long)blockIdx.z * LSEQ;

  if (tid < 64) { rowmax[tid] = -1e30f; rowsum[tid] = 0.f; }

  // Q fragments for this wave (rows q0 + wave*16 + lane%16), Dh in 2 chunks of 32
  v16bf qf[2];
  {
    const bf16* qrow = Q + (rowbase + q0 + wave * 16 + (lane & 15)) * DMODEL + colbase;
    const int kb = (lane >> 4) * 8;
#pragma unroll
    for (int c = 0; c < 2; ++c) {
      ((v8bf*)&qf[c])[0] = *(const v8bf*)(qrow + c * 32 + kb);
      ((v8bf*)&qf[c])[1] = *(const v8bf*)(qrow + c * 32 + kb + 16);
    }
  }

  v8f oacc[4] = {};   // 16 q-rows x 64 dh per wave

  for (int kt = 0; kt < LSEQ; kt += 64) {
    __syncthreads();   // Vt/Pbuf from previous iteration no longer in use

    // stage V tile transposed into LDS: issue all 4 loads first, then scatter
    {
      v8bf vv[4];
      int  vr[4], vg[4];
#pragma unroll
      for (int j = 0; j < 4; ++j) {
        const int i = tid + j * 128;
        vr[j] = i >> 3;              // key within tile
        vg[j] = (i & 7) * 8;         // dh group
        vv[j] = *(const v8bf*)(V + (rowbase + kt + vr[j]) * DMODEL + colbase + vg[j]);
      }
#pragma unroll
      for (int j = 0; j < 4; ++j)
#pragma unroll
        for (int e = 0; e < 8; ++e) Vt[(vg[j] + e) * 64 + vr[j]] = vv[j][e];
    }

    // S = Q @ K^T for this wave's 16 rows x 64 keys
    v8f sacc[4] = {};
#pragma unroll
    for (int c = 0; c < 2; ++c) {
#pragma unroll
      for (int ni = 0; ni < 4; ++ni) {
        const bf16* krow = K + (rowbase + kt + ni * 16 + (lane & 15)) * DMODEL +
                           colbase + c * 32 + (lane >> 4) * 8;
        v16bf bfv;
        ((v8bf*)&bfv)[0] = *(const v8bf*)(krow);
        ((v8bf*)&bfv)[1] = *(const v8bf*)(krow + 16);
        sacc[ni] = __builtin_amdgcn_wmma_f32_16x16x32_bf16(
            false, qf[c], false, bfv, (short)0, sacc[ni], false, false);
      }
    }
#pragma unroll
    for (int ni = 0; ni < 4; ++ni) {
      const int colp  = ni * 16 + (lane & 15);
      const int rbase = wave * 16 + (lane >> 4) * 8;
#pragma unroll
      for (int i = 0; i < 8; ++i) Sbuf[(rbase + i) * 64 + colp] = sacc[ni][i];
    }
    __syncthreads();

    // online softmax (one thread per q-row)
    if (tid < 64) {
      float m = rowmax[tid], mt = m;
      float* srow = &Sbuf[tid * 64];
#pragma unroll 8
      for (int j = 0; j < 64; ++j) { float s = srow[j] * 0.125f; srow[j] = s; mt = fmaxf(mt, s); }
      const float sc = __expf(m - mt);
      float sum = rowsum[tid] * sc;
      bf16* prow = &Pbuf[tid * 64];
#pragma unroll 8
      for (int j = 0; j < 64; ++j) { float p = __expf(srow[j] - mt); sum += p; prow[j] = f2bf(p); }
      rowmax[tid] = mt; rowsum[tid] = sum; rowscale[tid] = sc;
    }
    __syncthreads();

    // rescale O, then O += P @ V
    {
      const int rbase = wave * 16 + (lane >> 4) * 8;
#pragma unroll
      for (int i = 0; i < 8; ++i) {
        const float sc = rowscale[rbase + i];
#pragma unroll
        for (int ni = 0; ni < 4; ++ni) oacc[ni][i] *= sc;
      }
#pragma unroll
      for (int c = 0; c < 2; ++c) {
        const bf16* prow = &Pbuf[(wave * 16 + (lane & 15)) * 64 + c * 32 + (lane >> 4) * 8];
        v16bf pf;
        ((v8bf*)&pf)[0] = *(const v8bf*)(prow);
        ((v8bf*)&pf)[1] = *(const v8bf*)(prow + 16);
#pragma unroll
        for (int ni = 0; ni < 4; ++ni) {
          const bf16* vrow = &Vt[(ni * 16 + (lane & 15)) * 64 + c * 32 + (lane >> 4) * 8];
          v16bf vf;
          ((v8bf*)&vf)[0] = *(const v8bf*)(vrow);
          ((v8bf*)&vf)[1] = *(const v8bf*)(vrow + 16);
          oacc[ni] = __builtin_amdgcn_wmma_f32_16x16x32_bf16(
              false, pf, false, vf, (short)0, oacc[ni], false, false);
        }
      }
    }
  }

  __syncthreads();
  {
    const int rbase = wave * 16 + (lane >> 4) * 8;
#pragma unroll
    for (int i = 0; i < 8; ++i) {
      const float inv = 1.0f / rowsum[rbase + i];
      const long orow = (rowbase + q0 + rbase + i) * DMODEL;
#pragma unroll
      for (int ni = 0; ni < 4; ++ni)
        O[orow + colbase + ni * 16 + (lane & 15)] = f2bf(oacc[ni][i] * inv);
    }
  }
}

// ---------------------------------------------------------------------------
extern "C" void kernel_launch(void* const* d_in, const int* in_sizes, int n_in,
                              void* d_out, int out_size, void* d_ws, size_t ws_size,
                              hipStream_t stream) {
  const float* a      = (const float*)d_in[0];
  const float* b      = (const float*)d_in[1];
  const float* ln_a_g = (const float*)d_in[2];
  const float* ln_a_b = (const float*)d_in[3];
  const float* ln_b_g = (const float*)d_in[4];
  const float* ln_b_b = (const float*)d_in[5];
  const float* W_q  = (const float*)d_in[6];  const float* b_q  = (const float*)d_in[7];
  const float* W_k  = (const float*)d_in[8];  const float* b_k  = (const float*)d_in[9];
  const float* W_va = (const float*)d_in[10]; const float* b_va = (const float*)d_in[11];
  const float* W_vb = (const float*)d_in[12]; const float* b_vb = (const float*)d_in[13];
  const float* W_oa = (const float*)d_in[14]; const float* b_oa = (const float*)d_in[15];
  const float* W_ob = (const float*)d_in[16]; const float* b_ob = (const float*)d_in[17];

  char* ws = (char*)d_ws;
  size_t off = 0;
  auto carve = [&](size_t bytes) -> void* {
    void* p = ws + off;
    off += (bytes + 255) & ~(size_t)255;
    return p;
  };
  const size_t szW = (size_t)DMODEL * DMODEL * sizeof(bf16);  // 2 MB
  const size_t szX = (size_t)MROWS  * DMODEL * sizeof(bf16);  // 16 MB

  bf16* Wt_q  = (bf16*)carve(szW);
  bf16* Wt_k  = (bf16*)carve(szW);
  bf16* Wt_va = (bf16*)carve(szW);
  bf16* Wt_vb = (bf16*)carve(szW);
  bf16* Wt_oa = (bf16*)carve(szW);
  bf16* Wt_ob = (bf16*)carve(szW);
  bf16* an_h  = (bf16*)carve(szX);
  bf16* bn_h  = (bf16*)carve(szX);
  bf16* qh    = (bf16*)carve(szX);
  bf16* kh    = (bf16*)carve(szX);
  bf16* vah   = (bf16*)carve(szX);
  bf16* vbh   = (bf16*)carve(szX);
  bf16* oah   = (bf16*)carve(szX);
  bf16* obh   = (bf16*)carve(szX);

  float* out_a = (float*)d_out;
  float* out_b = (float*)d_out + (size_t)MROWS * DMODEL;

  // 1) weights -> bf16 transposed
  wcvt_kernel<<<dim3(DMODEL * DMODEL / (256 * 4), 6), 256, 0, stream>>>(
      W_q, W_k, W_va, W_vb, W_oa, W_ob, Wt_q, Wt_k, Wt_va, Wt_vb, Wt_oa, Wt_ob);

  // 2) layernorm -> bf16
  ln_kernel<<<dim3(MROWS, 2), 256, 0, stream>>>(
      a, b, ln_a_g, ln_a_b, ln_b_g, ln_b_b, an_h, bn_h);

  // 3) q/k/va/vb projections (WMMA GEMM, bf16 out)
  const dim3 gg(MROWS / 128, DMODEL / 128);
  gemm_kernel<false><<<gg, 256, 0, stream>>>(an_h, Wt_q,  b_q,  nullptr, qh,  nullptr, MROWS, DMODEL, DMODEL);
  gemm_kernel<false><<<gg, 256, 0, stream>>>(bn_h, Wt_k,  b_k,  nullptr, kh,  nullptr, MROWS, DMODEL, DMODEL);
  gemm_kernel<false><<<gg, 256, 0, stream>>>(an_h, Wt_va, b_va, nullptr, vah, nullptr, MROWS, DMODEL, DMODEL);
  gemm_kernel<false><<<gg, 256, 0, stream>>>(bn_h, Wt_vb, b_vb, nullptr, vbh, nullptr, MROWS, DMODEL, DMODEL);

  // 4) flash attention, both directions (attn_b is flash-attn with Q/K swapped)
  const dim3 ga(LSEQ / 64, NHEAD, BB);
  attn_kernel<<<ga, 128, 0, stream>>>(qh, kh, vbh, oah);  // out_a = softmax(QK^T) @ vb
  attn_kernel<<<ga, 128, 0, stream>>>(kh, qh, vah, obh);  // out_b = softmax(KQ^T) @ va

  // 5) output projections + bias + residual -> fp32 d_out
  gemm_kernel<true><<<gg, 256, 0, stream>>>(oah, Wt_oa, b_oa, a, nullptr, out_a, MROWS, DMODEL, DMODEL);
  gemm_kernel<true><<<gg, 256, 0, stream>>>(obh, Wt_ob, b_ob, b, nullptr, out_b, MROWS, DMODEL, DMODEL);
}
